// BindingSiteGNN_44908178047027
// MI455X (gfx1250) — matrix operand order repaired and checked by reading, not compiled, not run on hardware
//
#include <hip/hip_runtime.h>
#include <hip/hip_bf16.h>
#include <math.h>

typedef __attribute__((ext_vector_type(16))) _Float16 v16h;
typedef __attribute__((ext_vector_type(8)))  _Float16 v8h;
typedef __attribute__((ext_vector_type(8)))  float    v8f;

#define N_NODES 20000
#define E_EDGES 150000
#define E2      (E_EDGES + N_NODES)   // edges + self loops = 170000
#define IN_DIM  5
#define AA_EMBD 32
#define TOT_IN  37
#define K1PAD   64                    // 37 padded to multiple of 32
#define HID     128
#define HEADS   4
#define F1      (HEADS * HID)         // 512
#define OUT_DIM 2
#define NT      4                     // N-tiles (16 cols each) per wave in GEMM

// ---------- helpers ----------
__device__ __forceinline__ void atomicMaxF(float* addr, float val) {
  if (val >= 0.f)
    atomicMax((int*)addr, __float_as_int(val));
  else
    atomicMin((unsigned int*)addr, __float_as_uint(val));
}

__global__ void k_fill(float* __restrict__ p, float v, int n) {
  int i = blockIdx.x * blockDim.x + threadIdx.x;
  if (i < n) p[i] = v;
}

// ---------- preprocessing ----------
__global__ void k_deg_loopea(const int* __restrict__ dst, const float* __restrict__ ea,
                             float* __restrict__ deg, float* __restrict__ loop_ea) {
  int e = blockIdx.x * blockDim.x + threadIdx.x;
  if (e >= E_EDGES) return;
  int d = dst[e];
  atomicAdd(&deg[d], 1.f);
  atomicAdd(&loop_ea[d * 2 + 0], ea[e * 2 + 0]);
  atomicAdd(&loop_ea[d * 2 + 1], ea[e * 2 + 1]);
}

__global__ void k_loopea_div(float* __restrict__ loop_ea, const float* __restrict__ deg) {
  int n = blockIdx.x * blockDim.x + threadIdx.x;
  if (n >= N_NODES) return;
  float dg = deg[n];
  if (dg < 1.f) dg = 1.f;
  loop_ea[n * 2 + 0] /= dg;
  loop_ea[n * 2 + 1] /= dg;
}

__global__ void k_build_h0(const float* __restrict__ x, const int* __restrict__ rtype,
                           const float* __restrict__ aa_emb, _Float16* __restrict__ h0) {
  int idx = blockIdx.x * blockDim.x + threadIdx.x;
  if (idx >= N_NODES * K1PAD) return;
  int n = idx >> 6;
  int c = idx & 63;
  float v = 0.f;
  if (c < IN_DIM)       v = x[n * IN_DIM + c];
  else if (c < TOT_IN)  v = aa_emb[rtype[n] * AA_EMBD + (c - IN_DIM)];
  h0[idx] = (_Float16)v;
}

// ---------- pack f32 weight [Ksrc x Nn] into WMMA B-operand layout, f16 ----------
// Layout: Bp[((tn*(K/32) + kb)*32 + lane)*16 + j]
//   holds W[k, n] with k = kb*32 + (lane>>4)*16 + j,  n = tn*16 + (lane&15)
// so each lane's 16 elements for one (tile, k-block) are contiguous (32 B).
__global__ void k_pack_b(const float* __restrict__ W, _Float16* __restrict__ Bp,
                         int Ksrc, int K, int Nn) {
  int kbCount = K >> 5;
  int total = (Nn >> 4) * kbCount * 512;
  int idx = blockIdx.x * blockDim.x + threadIdx.x;
  if (idx >= total) return;
  int j  = idx & 15;
  int l  = (idx >> 4) & 31;
  int t  = idx >> 9;
  int kb = t % kbCount;
  int tn = t / kbCount;
  int k  = (kb << 5) + ((l >> 4) << 4) + j;
  int n  = (tn << 4) + (l & 15);
  Bp[idx] = (_Float16)((k < Ksrc) ? W[k * Nn + n] : 0.f);
}

// ---------- WMMA tiled GEMM: C[MxN] = A[MxK] * B[KxN] ----------
// A row-major f16, B pre-packed (k_pack_b), C f16 out (f32 accum).
// One wave computes a 16x64 tile (4 accumulators). M%16==0, N%64==0, K%32==0.
__global__ void k_wmma_gemm(const _Float16* __restrict__ A, const _Float16* __restrict__ Bp,
                            _Float16* __restrict__ C, int M, int K, int Nn) {
  const int wavesPerBlock = blockDim.x >> 5;
  int wave = blockIdx.x * wavesPerBlock + (threadIdx.x >> 5);
  int tilesM  = M >> 4;
  int groupsN = Nn >> 6;                    // groups of NT=4 tiles
  if (wave >= tilesM * groupsN) return;
  int tm  = wave / groupsN;
  int tg  = wave - tm * groupsN;
  int tn0 = tg * NT;
  int lane = threadIdx.x & 31;
  int half = lane >> 4;                     // 0: lanes 0-15, 1: lanes 16-31
  int l16  = lane & 15;
  int kbCount = K >> 5;

  v8f acc[NT] = {};
  const _Float16* Arow = A + (size_t)(tm * 16 + l16) * K;
  const _Float16* Blane = Bp + (size_t)lane * 16;

  for (int kb = 0; kb < kbCount; ++kb) {
    int k0 = kb << 5;
    // A 16x32 f16 operand: elems 0..15 -> K=half*16+j (matches packed B)
    v8h alo = *(const v8h*)(Arow + k0 + half * 16);
    v8h ahi = *(const v8h*)(Arow + k0 + half * 16 + 8);
    v16h a = __builtin_shufflevector(alo, ahi, 0, 1, 2, 3, 4, 5, 6, 7,
                                     8, 9, 10, 11, 12, 13, 14, 15);
#pragma unroll
    for (int t = 0; t < NT; ++t) {
      const _Float16* bp = Blane + (((size_t)(tn0 + t) * kbCount + kb) << 9);
      v8h blo = *(const v8h*)bp;
      v8h bhi = *(const v8h*)(bp + 8);
      v16h b = __builtin_shufflevector(blo, bhi, 0, 1, 2, 3, 4, 5, 6, 7,
                                       8, 9, 10, 11, 12, 13, 14, 15);
      acc[t] = __builtin_amdgcn_wmma_f32_16x16x32_f16(false, a, false, b,
                                                      (short)0, acc[t], false, false);
    }
  }
  // D layout: VGPR r -> M = r + 8*half, N = lane%16
#pragma unroll
  for (int t = 0; t < NT; ++t) {
    _Float16* Crow = C + (size_t)(tm * 16) * Nn + (tn0 + t) * 16 + l16;
#pragma unroll
    for (int r = 0; r < 8; ++r)
      Crow[(size_t)(r + 8 * half) * Nn] = (_Float16)acc[t][r];
  }
}

// ---------- layer 1 edge kernels (H=4, C=128) ----------
__global__ void k_alpha1(const int* __restrict__ src, const int* __restrict__ dst,
                         const float* __restrict__ ea, const float* __restrict__ loop_ea,
                         const _Float16* __restrict__ xl, const _Float16* __restrict__ xr,
                         const float* __restrict__ W1e, const float* __restrict__ att1,
                         float* __restrict__ alpha, float* __restrict__ amax) {
  __shared__ float s_att[F1];
  __shared__ float s_We[2 * F1];
  for (int i = threadIdx.x; i < F1; i += blockDim.x)     s_att[i] = att1[i];
  for (int i = threadIdx.x; i < 2 * F1; i += blockDim.x) s_We[i]  = W1e[i];
  __syncthreads();

  int e = blockIdx.x * (blockDim.x >> 5) + (threadIdx.x >> 5);
  if (e >= E2) return;
  int lane = threadIdx.x & 31;

  int s, d; float e0, e1;
  if (e < E_EDGES) { s = src[e]; d = dst[e]; e0 = ea[e * 2]; e1 = ea[e * 2 + 1]; }
  else             { s = d = e - E_EDGES;    e0 = loop_ea[s * 2]; e1 = loop_ea[s * 2 + 1]; }

  const _Float16* xls = xl + (size_t)s * F1;
  const _Float16* xrd = xr + (size_t)d * F1;

#pragma unroll
  for (int h = 0; h < HEADS; ++h) {
    float partial = 0.f;
#pragma unroll
    for (int j = 0; j < 4; ++j) {
      int c = h * HID + j * 32 + lane;                 // flat channel index
      float v = (float)xls[c] + (float)xrd[c] + e0 * s_We[c] + e1 * s_We[F1 + c];
      v = (v > 0.f) ? v : 0.2f * v;                    // leaky_relu(0.2)
      partial += v * s_att[c];
    }
    for (int off = 16; off > 0; off >>= 1) partial += __shfl_xor(partial, off, 32);
    if (lane == 0) {
      alpha[e * HEADS + h] = partial;
      atomicMaxF(&amax[d * HEADS + h], partial);
    }
  }
}

__global__ void k_softmax_denom1(const int* __restrict__ dst, float* __restrict__ alpha,
                                 const float* __restrict__ amax, float* __restrict__ denom) {
  int i = blockIdx.x * blockDim.x + threadIdx.x;
  if (i >= E2 * HEADS) return;
  int e = i >> 2, h = i & 3;
  int d = (e < E_EDGES) ? dst[e] : (e - E_EDGES);
  float p = expf(alpha[i] - amax[d * HEADS + h]);
  alpha[i] = p;
  atomicAdd(&denom[d * HEADS + h], p);
}

__global__ void k_aggregate1(const int* __restrict__ src, const int* __restrict__ dst,
                             const _Float16* __restrict__ xl, const float* __restrict__ alpha,
                             const float* __restrict__ denom, float* __restrict__ out1) {
  int e = blockIdx.x * (blockDim.x >> 5) + (threadIdx.x >> 5);
  if (e >= E2) return;
  int lane = threadIdx.x & 31;
  int s, d;
  if (e < E_EDGES) { s = src[e]; d = dst[e]; } else { s = d = e - E_EDGES; }
  float a[HEADS];
#pragma unroll
  for (int h = 0; h < HEADS; ++h) a[h] = alpha[e * HEADS + h] / denom[d * HEADS + h];
  const _Float16* xls = xl + (size_t)s * F1;
  float* od = out1 + (size_t)d * F1;
#pragma unroll
  for (int j = 0; j < 16; ++j) {
    int c = j * 32 + lane;
    atomicAdd(&od[c], (float)xls[c] * a[c >> 7]);
  }
}

__global__ void k_bias_elu_cvt(const float* __restrict__ acc, const float* __restrict__ bias,
                               _Float16* __restrict__ hout, int F, int total) {
  int i = blockIdx.x * blockDim.x + threadIdx.x;
  if (i >= total) return;
  int f = i % F;
  float v = acc[i] + bias[f];
  v = (v > 0.f) ? v : (expf(v) - 1.f);                 // ELU
  hout[i] = (_Float16)v;
}

// ---------- layer 2 edge kernels (H=1, C=128) ----------
__global__ void k_alpha2(const int* __restrict__ src, const int* __restrict__ dst,
                         const float* __restrict__ ea, const float* __restrict__ loop_ea,
                         const _Float16* __restrict__ xl, const _Float16* __restrict__ xr,
                         const float* __restrict__ W2e, const float* __restrict__ att2,
                         float* __restrict__ alpha, float* __restrict__ amax) {
  __shared__ float s_att[HID];
  __shared__ float s_We[2 * HID];
  for (int i = threadIdx.x; i < HID; i += blockDim.x)     s_att[i] = att2[i];
  for (int i = threadIdx.x; i < 2 * HID; i += blockDim.x) s_We[i]  = W2e[i];
  __syncthreads();

  int e = blockIdx.x * (blockDim.x >> 5) + (threadIdx.x >> 5);
  if (e >= E2) return;
  int lane = threadIdx.x & 31;

  int s, d; float e0, e1;
  if (e < E_EDGES) { s = src[e]; d = dst[e]; e0 = ea[e * 2]; e1 = ea[e * 2 + 1]; }
  else             { s = d = e - E_EDGES;    e0 = loop_ea[s * 2]; e1 = loop_ea[s * 2 + 1]; }

  const _Float16* xls = xl + (size_t)s * HID;
  const _Float16* xrd = xr + (size_t)d * HID;
  float partial = 0.f;
#pragma unroll
  for (int j = 0; j < 4; ++j) {
    int c = j * 32 + lane;
    float v = (float)xls[c] + (float)xrd[c] + e0 * s_We[c] + e1 * s_We[HID + c];
    v = (v > 0.f) ? v : 0.2f * v;
    partial += v * s_att[c];
  }
  for (int off = 16; off > 0; off >>= 1) partial += __shfl_xor(partial, off, 32);
  if (lane == 0) {
    alpha[e] = partial;
    atomicMaxF(&amax[d], partial);
  }
}

__global__ void k_softmax_denom2(const int* __restrict__ dst, float* __restrict__ alpha,
                                 const float* __restrict__ amax, float* __restrict__ denom) {
  int e = blockIdx.x * blockDim.x + threadIdx.x;
  if (e >= E2) return;
  int d = (e < E_EDGES) ? dst[e] : (e - E_EDGES);
  float p = expf(alpha[e] - amax[d]);
  alpha[e] = p;
  atomicAdd(&denom[d], p);
}

__global__ void k_aggregate2(const int* __restrict__ src, const int* __restrict__ dst,
                             const _Float16* __restrict__ xl, const float* __restrict__ alpha,
                             const float* __restrict__ denom, float* __restrict__ out2) {
  int e = blockIdx.x * (blockDim.x >> 5) + (threadIdx.x >> 5);
  if (e >= E2) return;
  int lane = threadIdx.x & 31;
  int s, d;
  if (e < E_EDGES) { s = src[e]; d = dst[e]; } else { s = d = e - E_EDGES; }
  float a = alpha[e] / denom[d];
  const _Float16* xls = xl + (size_t)s * HID;
  float* od = out2 + (size_t)d * HID;
#pragma unroll
  for (int j = 0; j < 4; ++j) {
    int c = j * 32 + lane;
    atomicAdd(&od[c], (float)xls[c] * a);
  }
}

// ---------- final projection: y = elu(out2 + b2) @ Wfc + bfc ----------
__global__ void k_final(const float* __restrict__ out2, const float* __restrict__ b2,
                        const float* __restrict__ Wfc, const float* __restrict__ bfc,
                        float* __restrict__ y) {
  int n = blockIdx.x * blockDim.x + threadIdx.x;
  if (n >= N_NODES) return;
  float a0 = bfc[0], a1 = bfc[1];
  const float* row = out2 + (size_t)n * HID;
#pragma unroll 4
  for (int c = 0; c < HID; ++c) {
    float v = row[c] + b2[c];
    v = (v > 0.f) ? v : (expf(v) - 1.f);
    a0 += v * Wfc[c * 2 + 0];
    a1 += v * Wfc[c * 2 + 1];
  }
  y[n * 2 + 0] = a0;
  y[n * 2 + 1] = a1;
}

// ---------- host ----------
extern "C" void kernel_launch(void* const* d_in, const int* in_sizes, int n_in,
                              void* d_out, int out_size, void* d_ws, size_t ws_size,
                              hipStream_t stream) {
  (void)in_sizes; (void)n_in; (void)out_size; (void)ws_size;
  const float* x      = (const float*)d_in[0];
  const int*   eidx   = (const int*)  d_in[1];
  const float* eattr  = (const float*)d_in[2];
  const int*   rtype  = (const int*)  d_in[3];
  const float* aa_emb = (const float*)d_in[4];
  const float* W1l    = (const float*)d_in[5];
  const float* W1r    = (const float*)d_in[6];
  const float* W1e    = (const float*)d_in[7];
  const float* att1   = (const float*)d_in[8];
  const float* b1     = (const float*)d_in[9];
  const float* W2l    = (const float*)d_in[10];
  const float* W2r    = (const float*)d_in[11];
  const float* W2e    = (const float*)d_in[12];
  const float* att2   = (const float*)d_in[13];
  const float* b2     = (const float*)d_in[14];
  const float* Wfc    = (const float*)d_in[15];
  const float* bfc    = (const float*)d_in[16];
  const int* src = eidx;
  const int* dst = eidx + E_EDGES;
  float* y = (float*)d_out;

  char* ws = (char*)d_ws;
  size_t off = 0;
  auto alloc = [&](size_t bytes) {
    size_t r = off;
    off += (bytes + 255) & ~(size_t)255;
    return r;
  };
  _Float16* h0      = (_Float16*)(ws + alloc((size_t)N_NODES * K1PAD * 2));
  _Float16* h1h     = (_Float16*)(ws + alloc((size_t)N_NODES * F1 * 2));
  _Float16* w1lp    = (_Float16*)(ws + alloc((size_t)K1PAD * F1 * 2));   // packed
  _Float16* w1rp    = (_Float16*)(ws + alloc((size_t)K1PAD * F1 * 2));
  _Float16* w2lp    = (_Float16*)(ws + alloc((size_t)F1 * HID * 2));
  _Float16* w2rp    = (_Float16*)(ws + alloc((size_t)F1 * HID * 2));
  _Float16* xlbuf   = (_Float16*)(ws + alloc((size_t)N_NODES * F1 * 2)); // xl1, reused as xl2
  _Float16* xrbuf   = (_Float16*)(ws + alloc((size_t)N_NODES * F1 * 2)); // xr1, reused as xr2
  float*    outbuf  = (float*)   (ws + alloc((size_t)N_NODES * F1 * 4)); // out1, reused as out2
  float*    deg     = (float*)   (ws + alloc((size_t)N_NODES * 4));
  float*    loop_ea = (float*)   (ws + alloc((size_t)N_NODES * 2 * 4));
  float*    alphab  = (float*)   (ws + alloc((size_t)E2 * HEADS * 4));   // reused for layer 2
  float*    amaxb   = (float*)   (ws + alloc((size_t)N_NODES * HEADS * 4));
  float*    denomb  = (float*)   (ws + alloc((size_t)N_NODES * HEADS * 4));

  auto cdiv = [](int a, int b) { return (a + b - 1) / b; };
  const int BLK = 256;
  const float NEG_INF = -__builtin_inff();

  // --- self-loop edge-attr means ---
  k_fill<<<cdiv(N_NODES, BLK), BLK, 0, stream>>>(deg, 0.f, N_NODES);
  k_fill<<<cdiv(2 * N_NODES, BLK), BLK, 0, stream>>>(loop_ea, 0.f, 2 * N_NODES);
  k_deg_loopea<<<cdiv(E_EDGES, BLK), BLK, 0, stream>>>(dst, eattr, deg, loop_ea);
  k_loopea_div<<<cdiv(N_NODES, BLK), BLK, 0, stream>>>(loop_ea, deg);

  // --- layer 1 input + packed weights ---
  k_build_h0<<<cdiv(N_NODES * K1PAD, BLK), BLK, 0, stream>>>(x, rtype, aa_emb, h0);
  k_pack_b<<<cdiv(K1PAD * F1, BLK), BLK, 0, stream>>>(W1l, w1lp, TOT_IN, K1PAD, F1);
  k_pack_b<<<cdiv(K1PAD * F1, BLK), BLK, 0, stream>>>(W1r, w1rp, TOT_IN, K1PAD, F1);

  // --- layer 1 GEMMs (WMMA): xl1 = h0 @ W1l, xr1 = h0 @ W1r ---
  {
    int tiles = (N_NODES / 16) * (F1 / 64);
    int blocks = cdiv(tiles, BLK / 32);
    k_wmma_gemm<<<blocks, BLK, 0, stream>>>(h0, w1lp, xlbuf, N_NODES, K1PAD, F1);
    k_wmma_gemm<<<blocks, BLK, 0, stream>>>(h0, w1rp, xrbuf, N_NODES, K1PAD, F1);
  }

  // --- layer 1 attention ---
  k_fill<<<cdiv(N_NODES * HEADS, BLK), BLK, 0, stream>>>(amaxb, NEG_INF, N_NODES * HEADS);
  k_fill<<<cdiv(N_NODES * HEADS, BLK), BLK, 0, stream>>>(denomb, 0.f, N_NODES * HEADS);
  k_alpha1<<<cdiv(E2, BLK / 32), BLK, 0, stream>>>(src, dst, eattr, loop_ea, xlbuf, xrbuf,
                                                   W1e, att1, alphab, amaxb);
  k_softmax_denom1<<<cdiv(E2 * HEADS, BLK), BLK, 0, stream>>>(dst, alphab, amaxb, denomb);
  k_fill<<<cdiv(N_NODES * F1, BLK), BLK, 0, stream>>>(outbuf, 0.f, N_NODES * F1);
  k_aggregate1<<<cdiv(E2, BLK / 32), BLK, 0, stream>>>(src, dst, xlbuf, alphab, denomb, outbuf);
  k_bias_elu_cvt<<<cdiv(N_NODES * F1, BLK), BLK, 0, stream>>>(outbuf, b1, h1h, F1, N_NODES * F1);

  // --- layer 2 packed weights + GEMMs ---
  k_pack_b<<<cdiv(F1 * HID, BLK), BLK, 0, stream>>>(W2l, w2lp, F1, F1, HID);
  k_pack_b<<<cdiv(F1 * HID, BLK), BLK, 0, stream>>>(W2r, w2rp, F1, F1, HID);
  {
    int tiles = (N_NODES / 16) * (HID / 64);
    int blocks = cdiv(tiles, BLK / 32);
    k_wmma_gemm<<<blocks, BLK, 0, stream>>>(h1h, w2lp, xlbuf, N_NODES, F1, HID);
    k_wmma_gemm<<<blocks, BLK, 0, stream>>>(h1h, w2rp, xrbuf, N_NODES, F1, HID);
  }

  // --- layer 2 attention (H=1) ---
  k_fill<<<cdiv(N_NODES, BLK), BLK, 0, stream>>>(amaxb, NEG_INF, N_NODES);
  k_fill<<<cdiv(N_NODES, BLK), BLK, 0, stream>>>(denomb, 0.f, N_NODES);
  k_alpha2<<<cdiv(E2, BLK / 32), BLK, 0, stream>>>(src, dst, eattr, loop_ea, xlbuf, xrbuf,
                                                   W2e, att2, alphab, amaxb);
  k_softmax_denom2<<<cdiv(E2, BLK), BLK, 0, stream>>>(dst, alphab, amaxb, denomb);
  k_fill<<<cdiv(N_NODES * HID, BLK), BLK, 0, stream>>>(outbuf, 0.f, N_NODES * HID);
  k_aggregate2<<<cdiv(E2, BLK / 32), BLK, 0, stream>>>(src, dst, xlbuf, alphab, denomb, outbuf);

  // --- final projection ---
  k_final<<<cdiv(N_NODES, BLK), BLK, 0, stream>>>(outbuf, b2, Wfc, bfc, y);
}
